// ContextualLoss_25417616457762
// MI455X (gfx1250) — matrix operand order repaired and checked by reading, not compiled, not run on hardware
//
#include <hip/hip_runtime.h>
#include <hip/hip_bf16.h>
#include <hip/hip_fp16.h>

// Contextual loss, B=8, C=128, N=4096 (64x64), fp32 in, fp32 scalar out.
// f16-normalized feature GEMM via v_wmma_f32_16x16x32_f16; opposite-side
// operand tiles staged workgroup-cooperatively into LDS via
// global_load_async_to_lds_b128 (ASYNCcnt) with double buffering, so the
// 8x4096x4096 distance matrix is never materialized and per-wave redundant
// L2 traffic is cut 8x.

typedef __attribute__((ext_vector_type(16))) _Float16 v16h;
typedef __attribute__((ext_vector_type(8)))  _Float16 v8h;
typedef __attribute__((ext_vector_type(8)))  float    v8f;
typedef __attribute__((ext_vector_type(4)))  int      v4i;

#define NB   8
#define NC   128
#define NPOS 4096
#define TILE_HALFS (16 * NC)   // 2048 f16 = 4 KB per 16-vector tile

#define AS1 __attribute__((address_space(1)))
#define AS3 __attribute__((address_space(3)))

#if defined(__has_builtin)
# if __has_builtin(__builtin_amdgcn_global_load_async_to_lds_b128)
#  define HAVE_ASYNC_LDS 1
# else
#  define HAVE_ASYNC_LDS 0
# endif
#else
# define HAVE_ASYNC_LDS 0
#endif

// ---- staging helpers --------------------------------------------------------

// Copy 16 bytes global -> LDS (one lane's share of a cooperative tile stage).
__device__ __forceinline__ void stage16(const _Float16* g, _Float16* l) {
#if HAVE_ASYNC_LDS
    __builtin_amdgcn_global_load_async_to_lds_b128(
        (AS1 v4i*)g, (AS3 v4i*)l, 0, 0);
#else
    *(float4*)l = *(const float4*)g;
#endif
}

template <int N>
__device__ __forceinline__ void wait_stage() {
#if HAVE_ASYNC_LDS
# if defined(__has_builtin) && __has_builtin(__builtin_amdgcn_s_wait_asynccnt)
    __builtin_amdgcn_s_wait_asynccnt((short)N);
# else
    asm volatile("s_wait_asynccnt %0" ::"n"(N) : "memory");
# endif
#endif
}

// ---- WMMA helpers -----------------------------------------------------------

// A-matrix 16x32 f16 fragment: lane half selects K chunk base
// kb = 16*(lane>>4); halves[0..15] = K kb..kb+15 (contiguous, matches B).
__device__ __forceinline__ v16h load_fragA(const _Float16* p) {
    v8h lo = *(const v8h*)(p);
    v8h hi = *(const v8h*)(p + 8);
    v16h f;
#pragma unroll
    for (int i = 0; i < 8; ++i) { f[i] = lo[i]; f[i + 8] = hi[i]; }
    return f;
}

// B-matrix 32x16 f16 fragment: lane holds column (lane&15), 16 contiguous K
// starting at 16*(lane>>4).
__device__ __forceinline__ v16h load_fragB(const _Float16* p) {
    return *(const v16h*)(p);
}

__device__ __forceinline__ v8f wmma_f16(v16h a, v16h b, v8f c) {
    return __builtin_amdgcn_wmma_f32_16x16x32_f16(
        /*neg_a=*/false, a, /*neg_b=*/false, b,
        /*c_mod=*/(short)0, c, /*reuse_a=*/false, /*reuse_b=*/false);
}

// ---- Kernel 1: per-channel mean of targets ---------------------------------

__global__ __launch_bounds__(256) void cx_mu_kernel(const float* __restrict__ tgt,
                                                    float* __restrict__ mu) {
    __shared__ float s[256];
    const int c = blockIdx.x;
    const int tid = threadIdx.x;
    float local = 0.0f;
    for (int b = 0; b < NB; ++b) {
        const float* p = tgt + ((size_t)b * NC + c) * NPOS;
        for (int j = tid; j < NPOS; j += 256) local += p[j];
    }
    s[tid] = local;
    __syncthreads();
    for (int st = 128; st > 0; st >>= 1) {
        if (tid < st) s[tid] += s[tid + st];
        __syncthreads();
    }
    if (tid == 0) mu[c] = s[0] * (1.0f / (NB * NPOS));
}

// ---- Kernel 2: center, L2-normalize over C, emit f16 [b][pos][c] -----------

__global__ __launch_bounds__(256) void cx_norm_kernel(const float* __restrict__ x,
                                                      const float* __restrict__ y,
                                                      const float* __restrict__ mu,
                                                      _Float16* __restrict__ Xh,
                                                      _Float16* __restrict__ Yh) {
    __shared__ float smu[NC];
    const int tid = threadIdx.x;
    if (tid < NC) smu[tid] = mu[tid];
    __syncthreads();
    const int idx = blockIdx.x * 256 + tid;      // 0..32767 = b*4096 + n
    const int b = idx >> 12;
    const int n = idx & (NPOS - 1);
    const size_t base = ((size_t)b * NC) * NPOS + n;
    float sx = 0.0f, sy = 0.0f;
#pragma unroll 4
    for (int c = 0; c < NC; ++c) {
        float xv = x[base + (size_t)c * NPOS] - smu[c];
        float yv = y[base + (size_t)c * NPOS] - smu[c];
        sx += xv * xv;
        sy += yv * yv;
    }
    const float rx = 1.0f / fmaxf(sqrtf(sx), 1e-12f);
    const float ry = 1.0f / fmaxf(sqrtf(sy), 1e-12f);
    _Float16* xo = Xh + (size_t)idx * NC;
    _Float16* yo = Yh + (size_t)idx * NC;
#pragma unroll 4
    for (int c = 0; c < NC; ++c) {
        float xv = (x[base + (size_t)c * NPOS] - smu[c]) * rx;
        float yv = (y[base + (size_t)c * NPOS] - smu[c]) * ry;
        xo[c] = (_Float16)xv;
        yo[c] = (_Float16)yv;
    }
}

// ---- Kernel 3: per-row dist-min then sum(w); emit A_i, C_i -----------------
// Block = 8 waves, all on batch b; each wave owns one 16-row tile and keeps its
// A-fragments in VGPRs. Y tiles are staged once per block into LDS (async,
// double-buffered) and swept twice: pass A = row-max(S), pass B = sum(w).
// rowA[i] = 2/(dmin_i + eps), rowC[i] = 2 - log(sum_j w_ij).

__global__ __launch_bounds__(256) void cx_rows_kernel(const _Float16* __restrict__ Xh,
                                                      const _Float16* __restrict__ Yh,
                                                      float* __restrict__ rowA,
                                                      float* __restrict__ rowC) {
    __shared__ __align__(16) _Float16 ytile[2][TILE_HALFS];
    const int tid = threadIdx.x;
    const int wave = (blockIdx.x * 256 + tid) >> 5;            // 0..2047
    const int lane = tid & 31;
    const int b = wave >> 8;                                    // 0..7 (uniform per block)
    const int n0 = (wave & 255) << 4;                           // row-tile base
    const int r = lane & 15;
    const int half = lane >> 4;
    const int kbA = half << 4;                                  // 0 or 16
    const int kbB = half << 4;                                  // 0 or 16

    const _Float16* xb = Xh + ((size_t)(b * NPOS + n0 + r)) * NC + kbA;
    const v16h a0 = load_fragA(xb);
    const v16h a1 = load_fragA(xb + 32);
    const v16h a2 = load_fragA(xb + 64);
    const v16h a3 = load_fragA(xb + 96);

    const _Float16* ysrc = Yh + (size_t)b * NPOS * NC;          // batch base
    const int so = tid * 8;                                     // this thread's 16B slot

    stage16(ysrc + so, &ytile[0][so]);                          // prologue: tile 0

    float rmax[8];
#pragma unroll
    for (int v = 0; v < 8; ++v) rmax[v] = -3.0e38f;

    // ---- Pass A: row max of similarity S (dist_min = 1 - max S) ----
    for (int mt = 0; mt < 256; ++mt) {
        const int nxt = (mt + 1) & 255;                         // mt=255 stages tile 0 for pass B
        stage16(ysrc + (size_t)nxt * TILE_HALFS + so, &ytile[(mt + 1) & 1][so]);
        wait_stage<1>();
        __syncthreads();
        const _Float16* t = &ytile[mt & 1][r * NC + kbB];
        v8f ca = {0.f, 0.f, 0.f, 0.f, 0.f, 0.f, 0.f, 0.f};
        v8f cb = {0.f, 0.f, 0.f, 0.f, 0.f, 0.f, 0.f, 0.f};
        ca = wmma_f16(a0, load_fragB(t),      ca);
        cb = wmma_f16(a1, load_fragB(t + 32), cb);
        ca = wmma_f16(a2, load_fragB(t + 64), ca);
        cb = wmma_f16(a3, load_fragB(t + 96), cb);
#pragma unroll
        for (int v = 0; v < 8; ++v) rmax[v] = fmaxf(rmax[v], ca[v] + cb[v]);
        __syncthreads();
    }
#pragma unroll
    for (int v = 0; v < 8; ++v) {
        rmax[v] = fmaxf(rmax[v], __shfl_xor(rmax[v], 1));
        rmax[v] = fmaxf(rmax[v], __shfl_xor(rmax[v], 2));
        rmax[v] = fmaxf(rmax[v], __shfl_xor(rmax[v], 4));
        rmax[v] = fmaxf(rmax[v], __shfl_xor(rmax[v], 8));
    }
    float inv[8], wsum[8];
#pragma unroll
    for (int v = 0; v < 8; ++v) {
        const float dmin = 1.0f - rmax[v];
        inv[v] = 2.0f / (dmin + 1e-5f);   // 1/(h*(dmin+eps)), h=0.5
        wsum[v] = 0.0f;
    }

    // ---- Pass B: row sums of w = exp(2 - d*inv); tile 0 already staged ----
    for (int mt = 0; mt < 256; ++mt) {
        if (mt != 255) {
            stage16(ysrc + (size_t)(mt + 1) * TILE_HALFS + so, &ytile[(mt + 1) & 1][so]);
            wait_stage<1>();
        } else {
            wait_stage<0>();
        }
        __syncthreads();
        const _Float16* t = &ytile[mt & 1][r * NC + kbB];
        v8f ca = {0.f, 0.f, 0.f, 0.f, 0.f, 0.f, 0.f, 0.f};
        v8f cb = {0.f, 0.f, 0.f, 0.f, 0.f, 0.f, 0.f, 0.f};
        ca = wmma_f16(a0, load_fragB(t),      ca);
        cb = wmma_f16(a1, load_fragB(t + 32), cb);
        ca = wmma_f16(a2, load_fragB(t + 64), ca);
        cb = wmma_f16(a3, load_fragB(t + 96), cb);
#pragma unroll
        for (int v = 0; v < 8; ++v) {
            const float d = 1.0f - (ca[v] + cb[v]);
            wsum[v] += __expf(2.0f - d * inv[v]);
        }
        __syncthreads();
    }
#pragma unroll
    for (int v = 0; v < 8; ++v) {
        wsum[v] += __shfl_xor(wsum[v], 1);
        wsum[v] += __shfl_xor(wsum[v], 2);
        wsum[v] += __shfl_xor(wsum[v], 4);
        wsum[v] += __shfl_xor(wsum[v], 8);
    }
    if ((lane & 15) == 0) {
        const int rbase = b * NPOS + n0 + (half << 3);
#pragma unroll
        for (int v = 0; v < 8; ++v) {
            rowA[rbase + v] = inv[v];
            rowC[rbase + v] = 2.0f - logf(wsum[v]);
        }
    }
}

// ---- Kernel 4: column max of w_ij/rowsum_i = exp(max_i (C_i - d_ij*A_i)) ---
// Block = 8 waves on batch b; each wave owns a 16-column tile (B-frags
// resident); X row tiles staged once per block into LDS, async double-buffered.

__global__ __launch_bounds__(256) void cx_cols_kernel(const _Float16* __restrict__ Xh,
                                                      const _Float16* __restrict__ Yh,
                                                      const float* __restrict__ rowA,
                                                      const float* __restrict__ rowC,
                                                      float* __restrict__ colmax) {
    __shared__ __align__(16) _Float16 xtile[2][TILE_HALFS];
    const int tid = threadIdx.x;
    const int wave = (blockIdx.x * 256 + tid) >> 5;            // 0..2047
    const int lane = tid & 31;
    const int b = wave >> 8;
    const int m0 = (wave & 255) << 4;                           // column-tile base
    const int r = lane & 15;
    const int half = lane >> 4;
    const int kbA = half << 4;
    const int kbB = half << 4;

    const _Float16* yb = Yh + ((size_t)(b * NPOS + m0 + r)) * NC + kbB;
    const v16h b0 = load_fragB(yb);
    const v16h b1 = load_fragB(yb + 32);
    const v16h b2 = load_fragB(yb + 64);
    const v16h b3 = load_fragB(yb + 96);

    const _Float16* xsrc = Xh + (size_t)b * NPOS * NC;
    const int so = tid * 8;
    const int rowoff = b * NPOS + (half << 3);

    stage16(xsrc + so, &xtile[0][so]);                          // prologue: tile 0

    float cm = -3.0e38f;
    for (int nt = 0; nt < 256; ++nt) {
        if (nt != 255) {
            stage16(xsrc + (size_t)(nt + 1) * TILE_HALFS + so, &xtile[(nt + 1) & 1][so]);
            wait_stage<1>();
        } else {
            wait_stage<0>();
        }
        __syncthreads();
        const _Float16* t = &xtile[nt & 1][r * NC + kbA];
        v8f ca = {0.f, 0.f, 0.f, 0.f, 0.f, 0.f, 0.f, 0.f};
        v8f cb = {0.f, 0.f, 0.f, 0.f, 0.f, 0.f, 0.f, 0.f};
        ca = wmma_f16(load_fragA(t),      b0, ca);
        cb = wmma_f16(load_fragA(t + 32), b1, cb);
        ca = wmma_f16(load_fragA(t + 64), b2, ca);
        cb = wmma_f16(load_fragA(t + 96), b3, cb);
        const float4 A0 = *(const float4*)(rowA + rowoff + nt * 16);
        const float4 A1 = *(const float4*)(rowA + rowoff + nt * 16 + 4);
        const float4 C0 = *(const float4*)(rowC + rowoff + nt * 16);
        const float4 C1 = *(const float4*)(rowC + rowoff + nt * 16 + 4);
        const float Av[8] = {A0.x, A0.y, A0.z, A0.w, A1.x, A1.y, A1.z, A1.w};
        const float Cv[8] = {C0.x, C0.y, C0.z, C0.w, C1.x, C1.y, C1.z, C1.w};
#pragma unroll
        for (int v = 0; v < 8; ++v) {
            const float d = 1.0f - (ca[v] + cb[v]);
            cm = fmaxf(cm, Cv[v] - d * Av[v]);
        }
        __syncthreads();
    }
    cm = fmaxf(cm, __shfl_xor(cm, 16));   // lanes l and l+16 share a column
    if (lane < 16) colmax[b * NPOS + m0 + lane] = __expf(cm);
}

// ---- Kernel 5: cx = mean_j colmax, loss = mean_b(-log(cx+eps)) -------------

__global__ __launch_bounds__(256) void cx_finalize_kernel(const float* __restrict__ colmax,
                                                          float* __restrict__ out) {
    __shared__ float s[256];
    const int tid = threadIdx.x;
    float acc = 0.0f;
    for (int b = 0; b < NB; ++b) {
        float local = 0.0f;
        for (int j = tid; j < NPOS; j += 256) local += colmax[b * NPOS + j];
        s[tid] = local;
        __syncthreads();
        for (int st = 128; st > 0; st >>= 1) {
            if (tid < st) s[tid] += s[tid + st];
            __syncthreads();
        }
        if (tid == 0) acc += -logf(s[0] * (1.0f / NPOS) + 1e-5f);
        __syncthreads();
    }
    if (tid == 0) out[0] = acc * (1.0f / NB);
}

// ---- Launch ----------------------------------------------------------------

extern "C" void kernel_launch(void* const* d_in, const int* in_sizes, int n_in,
                              void* d_out, int out_size, void* d_ws, size_t ws_size,
                              hipStream_t stream) {
    (void)in_sizes; (void)n_in; (void)out_size; (void)ws_size;
    const float* x = (const float*)d_in[0];   // inputs  [8,128,64,64]
    const float* y = (const float*)d_in[1];   // targets [8,128,64,64]
    float* out = (float*)d_out;

    // Workspace layout (~17.2 MB total)
    float* mu     = (float*)d_ws;             // 128 (padded to 256)
    float* rowA   = mu + 256;                 // 32768
    float* rowC   = rowA + 32768;             // 32768
    float* colmax = rowC + 32768;             // 32768
    _Float16* Xh  = (_Float16*)(colmax + 32768);            // 8*4096*128 f16
    _Float16* Yh  = Xh + (size_t)NB * NPOS * NC;            // 8*4096*128 f16

    cx_mu_kernel<<<128, 256, 0, stream>>>(y, mu);
    cx_norm_kernel<<<128, 256, 0, stream>>>(x, y, mu, Xh, Yh);
    cx_rows_kernel<<<256, 256, 0, stream>>>(Xh, Yh, rowA, rowC);
    cx_cols_kernel<<<256, 256, 0, stream>>>(Xh, Yh, rowA, rowC, colmax);
    cx_finalize_kernel<<<1, 256, 0, stream>>>(colmax, out);
}
